// CLIP_loss_72481868087647
// MI455X (gfx1250) — matrix operand order repaired and checked by reading, not compiled, not run on hardware
//
#include <hip/hip_runtime.h>
#include <hip/hip_bf16.h>

typedef __attribute__((ext_vector_type(16))) __bf16 v16bf;
typedef __attribute__((ext_vector_type(8)))  float  v8f;

// ---------- helpers ----------
__device__ __forceinline__ unsigned short f2bf_bits(float f) {
    unsigned u = __builtin_bit_cast(unsigned, f);
    unsigned r = u + 0x7FFFu + ((u >> 16) & 1u);   // round-to-nearest-even
    return (unsigned short)(r >> 16);
}
__device__ __forceinline__ __bf16 f2bf(float f) {
    unsigned short h = f2bf_bits(f);
    return __builtin_bit_cast(__bf16, h);
}
__device__ __forceinline__ float bf2f(__bf16 h) {
    unsigned short u = __builtin_bit_cast(unsigned short, h);
    unsigned x = ((unsigned)u) << 16;
    return __builtin_bit_cast(float, x);
}

union AB16 { v16bf v; uint4 q[2]; };

// ---------- fp32 -> bf16 conversion (vector x4) ----------
__global__ __launch_bounds__(256)
void f2bf4_kernel(const float4* __restrict__ in, __bf16* __restrict__ out, int n4) {
    int i = blockIdx.x * 256 + threadIdx.x;
    if (i >= n4) return;
    float4 v = in[i];
    __bf16* p = out + (size_t)i * 4;
    p[0] = f2bf(v.x); p[1] = f2bf(v.y); p[2] = f2bf(v.z); p[3] = f2bf(v.w);
}

// ---------- GEMM: C[M,N] = X[M,K](bf16) @ W[N,K]^T(bf16), fp32 accum ----------
// 256 threads = 8 waves; wave grid 4(M) x 2(N); block tile 64(M) x 128(N);
// each wave computes 16x64 via 4 accumulators of v_wmma_f32_16x16x32_bf16.
template <bool OUT_BF16>
__global__ __launch_bounds__(256)
void gemm_xwt_kernel(const __bf16* __restrict__ X, const __bf16* __restrict__ W,
                     void* __restrict__ Cout, int M, int N, int K) {
    const int lane  = threadIdx.x & 31;
    const int wave  = threadIdx.x >> 5;
    const int mBase = blockIdx.x * 64 + (wave & 3) * 16;
    const int nBase = blockIdx.y * 128 + (wave >> 2) * 64;

    const int mRow = mBase + (lane & 15);
    const int aKh  = (lane >> 4) * 8;     // A: lanes 0-15 -> K 0..7/16..23, 16-31 -> 8..15/24..31
    const int bN   = (lane & 15);
    const int bKh  = (lane >> 4) * 8;     // B: lanes 0-15 -> K 0..7/16..23, 16-31 -> 8..15/24..31

    v8f acc[4];
#pragma unroll
    for (int t = 0; t < 4; ++t) acc[t] = (v8f){0.f,0.f,0.f,0.f,0.f,0.f,0.f,0.f};

    const __bf16* aRow = X + (size_t)mRow * K + aKh;
    for (int k0 = 0; k0 < K; k0 += 32) {
        AB16 a;
        a.q[0] = *(const uint4*)(aRow + k0);
        a.q[1] = *(const uint4*)(aRow + k0 + 16);
#pragma unroll
        for (int t = 0; t < 4; ++t) {
            const __bf16* bRow = W + (size_t)(nBase + t * 16 + bN) * K + k0 + bKh;
            AB16 b;
            b.q[0] = *(const uint4*)(bRow);
            b.q[1] = *(const uint4*)(bRow + 16);
            acc[t] = __builtin_amdgcn_wmma_f32_16x16x32_bf16(
                false, a.v, false, b.v, (short)0, acc[t], false, false);
        }
    }

    const int cmOff = (lane >> 4) * 8;
    const int cn0   = nBase + (lane & 15);
#pragma unroll
    for (int t = 0; t < 4; ++t) {
#pragma unroll
        for (int v = 0; v < 8; ++v) {
            size_t idx = (size_t)(mBase + v + cmOff) * N + (cn0 + t * 16);
            if (OUT_BF16) ((__bf16*)Cout)[idx] = f2bf(acc[t][v]);
            else          ((float*)Cout)[idx]  = acc[t][v];
        }
    }
}

// ---------- fused LayerNorm + L2 normalize, one 512-wide row per block ----------
__device__ __forceinline__ float block_sum256(float v, float* red) {
    int tid = threadIdx.x;
    red[tid] = v; __syncthreads();
    for (int s = 128; s > 0; s >>= 1) {
        if (tid < s) red[tid] += red[tid + s];
        __syncthreads();
    }
    float r = red[0]; __syncthreads();
    return r;
}

__global__ __launch_bounds__(256)
void ln_l2_kernel(const float* __restrict__ H, const float* __restrict__ g,
                  const float* __restrict__ b, __bf16* __restrict__ Z) {
    __shared__ float red[256];
    const int D = 512;
    const int tid = threadIdx.x;
    const float* x = H + (size_t)blockIdx.x * D;
    float x0 = x[tid], x1 = x[tid + 256];

    float mu  = block_sum256(x0 + x1, red) * (1.0f / D);
    float d0 = x0 - mu, d1 = x1 - mu;
    float var = block_sum256(d0 * d0 + d1 * d1, red) * (1.0f / D);
    float inv = rsqrtf(var + 1e-5f);
    float y0 = d0 * inv * g[tid]       + b[tid];
    float y1 = d1 * inv * g[tid + 256] + b[tid + 256];
    float n2 = block_sum256(y0 * y0 + y1 * y1, red);
    float rn = 1.0f / fmaxf(sqrtf(n2), 1e-12f);
    __bf16* z = Z + (size_t)blockIdx.x * D;
    z[tid]       = f2bf(y0 * rn);
    z[tid + 256] = f2bf(y1 * rn);
}

// ---------- online row log-sum-exp of L = scale * ZA @ ZB^T ----------
// 128 threads = 4 waves, each wave owns 16 rows (block: 64 rows).
// Block's 64 A-rows staged into LDS with CDNA5 async global->LDS copies
// (ASYNCcnt path), then swept across all 8192 columns with WMMA while
// prefetching the next B chunk (global_prefetch).
__global__ __launch_bounds__(128)
void rowstats_kernel(const __bf16* __restrict__ ZA, const __bf16* __restrict__ ZB,
                     float* __restrict__ logZ, const float* __restrict__ lscale,
                     int Btot) {
    constexpr int D = 512;
    __shared__ __bf16 sA[64 * D];   // 64 KB

    const float scale = __expf(lscale[0]);
    const int lane = threadIdx.x & 31;
    const int wave = threadIdx.x >> 5;
    const int mBlock = blockIdx.x * 64;
    const int mBase  = mBlock + wave * 16;

    // stage 64 rows of ZA into LDS via async tensor-free DMA path
    // (global_load_async_to_lds_b128, tracked by ASYNCcnt)
    {
        const char* srcBase = (const char*)(ZA + (size_t)mBlock * D);
        unsigned ldsBase = (unsigned)(size_t)(&sA[0]);   // low 32 bits = LDS offset
        for (int i = threadIdx.x; i < 64 * D * 2 / 16; i += 128) {
            unsigned ldsOff = ldsBase + (unsigned)i * 16u;
            const char* g = srcBase + (size_t)i * 16;
            asm volatile("global_load_async_to_lds_b128 %0, %1, off"
                         :: "v"(ldsOff), "v"(g) : "memory");
        }
        asm volatile("s_wait_asynccnt 0x0" ::: "memory");
    }
    __syncthreads();

    const int aKh = (lane >> 4) * 8;
    const int bN  = (lane & 15);
    const int bKh = (lane >> 4) * 8;
    const __bf16* aRow = sA + (size_t)(wave * 16 + (lane & 15)) * D + aKh;

    float rm[8], rs[8];
#pragma unroll
    for (int v = 0; v < 8; ++v) { rm[v] = -1e30f; rs[v] = 0.0f; }

    for (int col = 0; col < Btot; col += 64) {
        // prefetch next column chunk's B rows into cache ahead of the WMMA sweep
        if (col + 64 < Btot) {
#pragma unroll
            for (int t = 0; t < 4; ++t)
                __builtin_prefetch(ZB + (size_t)(col + 64 + t * 16 + bN) * D + bKh, 0, 1);
        }

        v8f acc[4];
#pragma unroll
        for (int t = 0; t < 4; ++t) acc[t] = (v8f){0.f,0.f,0.f,0.f,0.f,0.f,0.f,0.f};

        for (int k0 = 0; k0 < D; k0 += 32) {
            AB16 a;
            a.q[0] = *(const uint4*)(aRow + k0);        // ds_load_b128 (or reg-promoted)
            a.q[1] = *(const uint4*)(aRow + k0 + 16);
#pragma unroll
            for (int t = 0; t < 4; ++t) {
                const __bf16* bRow = ZB + (size_t)(col + t * 16 + bN) * D + k0 + bKh;
                AB16 b;
                b.q[0] = *(const uint4*)(bRow);
                b.q[1] = *(const uint4*)(bRow + 16);
                acc[t] = __builtin_amdgcn_wmma_f32_16x16x32_bf16(
                    false, a.v, false, b.v, (short)0, acc[t], false, false);
            }
        }
        // online (max, sumexp) merge of this 64-column chunk
#pragma unroll
        for (int v = 0; v < 8; ++v) {
            float mx = rm[v];
#pragma unroll
            for (int t = 0; t < 4; ++t) mx = fmaxf(mx, scale * acc[t][v]);
            float s = rs[v] * __expf(rm[v] - mx);
#pragma unroll
            for (int t = 0; t < 4; ++t) s += __expf(scale * acc[t][v] - mx);
            rm[v] = mx; rs[v] = s;
        }
    }

    // reduce over the 16 lanes that share each row (within each half-wave)
    for (int d = 1; d < 16; d <<= 1) {
#pragma unroll
        for (int v = 0; v < 8; ++v) {
            float om = __shfl_xor(rm[v], d, 32);
            float os = __shfl_xor(rs[v], d, 32);
            float mx = fmaxf(rm[v], om);
            rs[v] = rs[v] * __expf(rm[v] - mx) + os * __expf(om - mx);
            rm[v] = mx;
        }
    }
    if ((lane & 15) == 0) {
        int hi = lane >> 4;
#pragma unroll
        for (int v = 0; v < 8; ++v)
            logZ[mBase + v + hi * 8] = rm[v] + __logf(rs[v]);
    }
}

// ---------- diagonal: diag[i] = scale * dot(z1[i], z2[i]) ----------
__global__ __launch_bounds__(256)
void diag_kernel(const __bf16* __restrict__ Z1, const __bf16* __restrict__ Z2,
                 float* __restrict__ diag, const float* __restrict__ lscale) {
    constexpr int D = 512;
    const float scale = __expf(lscale[0]);
    const int lane = threadIdx.x & 31;
    const int row  = blockIdx.x * 8 + (threadIdx.x >> 5);
    const __bf16* a = Z1 + (size_t)row * D + lane * 16;
    const __bf16* b = Z2 + (size_t)row * D + lane * 16;
    float s = 0.0f;
#pragma unroll
    for (int j = 0; j < 16; ++j) s += bf2f(a[j]) * bf2f(b[j]);
    for (int d = 16; d > 0; d >>= 1) s += __shfl_xor(s, d, 32);
    if (lane == 0) diag[row] = scale * s;
}

// ---------- final loss reduction ----------
__global__ __launch_bounds__(256)
void loss_kernel(const float* __restrict__ diag, const float* __restrict__ logZrow,
                 const float* __restrict__ logZcol, float* __restrict__ out, int Btot) {
    __shared__ float red[256];
    float s = 0.0f;
    for (int i = threadIdx.x; i < Btot; i += 256)
        s += (logZrow[i] - diag[i]) + (logZcol[i] - diag[i]);
    float tot = block_sum256(s, red);
    if (threadIdx.x == 0) out[0] = tot / (2.0f * (float)Btot);
}

// ---------- launch ----------
extern "C" void kernel_launch(void* const* d_in, const int* in_sizes, int n_in,
                              void* d_out, int out_size, void* d_ws, size_t ws_size,
                              hipStream_t stream) {
    (void)in_sizes; (void)n_in; (void)out_size; (void)ws_size;
    constexpr int Bn = 8192, LAT = 1024, J = 512;

    const float* latent[2] = {(const float*)d_in[0], (const float*)d_in[1]};
    const float* W1[2]     = {(const float*)d_in[2], (const float*)d_in[4]};
    const float* W2[2]     = {(const float*)d_in[3], (const float*)d_in[5]};
    const float* gamma[2]  = {(const float*)d_in[6], (const float*)d_in[8]};
    const float* beta[2]   = {(const float*)d_in[7], (const float*)d_in[9]};
    const float* lscale    = (const float*)d_in[10];

    char* ws = (char*)d_ws;
    size_t off = 0;
    auto alloc = [&](size_t bytes) -> void* {
        void* p = ws + off;
        off += (bytes + 255) & ~(size_t)255;
        return p;
    };
    __bf16* Zbf[2] = {(__bf16*)alloc((size_t)Bn * J * 2),
                      (__bf16*)alloc((size_t)Bn * J * 2)};
    float* logZrow = (float*)alloc(Bn * 4);
    float* logZcol = (float*)alloc(Bn * 4);
    float* diag    = (float*)alloc(Bn * 4);
    __bf16* Lbf  = (__bf16*)alloc((size_t)Bn * LAT * 2);
    __bf16* W1bf = (__bf16*)alloc((size_t)J * LAT * 2);
    __bf16* W2bf = (__bf16*)alloc((size_t)J * J * 2);
    __bf16* H1bf = (__bf16*)alloc((size_t)Bn * J * 2);
    float*  H2f  = (float*)alloc((size_t)Bn * J * 4);

    for (int s = 0; s < 2; ++s) {
        int n4;
        n4 = Bn * LAT / 4;
        f2bf4_kernel<<<(n4 + 255) / 256, 256, 0, stream>>>((const float4*)latent[s], Lbf, n4);
        n4 = J * LAT / 4;
        f2bf4_kernel<<<(n4 + 255) / 256, 256, 0, stream>>>((const float4*)W1[s], W1bf, n4);
        n4 = J * J / 4;
        f2bf4_kernel<<<(n4 + 255) / 256, 256, 0, stream>>>((const float4*)W2[s], W2bf, n4);

        gemm_xwt_kernel<true><<<dim3(Bn / 64, J / 128), 256, 0, stream>>>(
            Lbf, W1bf, (void*)H1bf, Bn, J, LAT);
        gemm_xwt_kernel<false><<<dim3(Bn / 64, J / 128), 256, 0, stream>>>(
            H1bf, W2bf, (void*)H2f, Bn, J, J);
        ln_l2_kernel<<<Bn, 256, 0, stream>>>(H2f, gamma[s], beta[s], Zbf[s]);
    }

    rowstats_kernel<<<Bn / 64, 128, 0, stream>>>(Zbf[0], Zbf[1], logZrow, lscale, Bn);
    rowstats_kernel<<<Bn / 64, 128, 0, stream>>>(Zbf[1], Zbf[0], logZcol, lscale, Bn);
    diag_kernel<<<Bn / 8, 256, 0, stream>>>(Zbf[0], Zbf[1], diag, lscale);
    loss_kernel<<<1, 256, 0, stream>>>(diag, logZrow, logZcol, (float*)d_out, Bn);
}
